// LSTM_74182675136807
// MI455X (gfx1250) — matrix operand order repaired and checked by reading, not compiled, not run on hardware
//
#include <hip/hip_runtime.h>

// ---- types matching gfx1250 WMMA builtin signatures ----
typedef __attribute__((ext_vector_type(16))) __bf16 v16bf;
typedef __attribute__((ext_vector_type(8)))  __bf16 v8bf;
typedef __attribute__((ext_vector_type(8)))  float  v8f;

#define B_TOT  1024
#define T_LEN  256
#define HIDN   256
#define EMBD   10
#define CLS    10
#define KAUG   288            // 256 (hidden) + 32 (emb padded 10->32)
#define MTILE  32             // batch rows per workgroup
#define NWG    (B_TOT / MTILE)

// -------------------------------------------------------------------------
// Prep: build augmented, transposed, bf16 weight matrix in workspace:
//   wcat[n][k], n in [0,1024) over gates {g,i,f,o}, k in [0,288)
// -------------------------------------------------------------------------
__global__ __launch_bounds__(256, 1)
void prep_weights(const float* __restrict__ Wgx, const float* __restrict__ Wgh,
                  const float* __restrict__ Wix, const float* __restrict__ Wih,
                  const float* __restrict__ Wfx, const float* __restrict__ Wfh,
                  const float* __restrict__ Wox, const float* __restrict__ Woh,
                  __bf16* __restrict__ wcat) {
  int idx = blockIdx.x * blockDim.x + threadIdx.x;
  if (idx >= 1024 * KAUG) return;
  int n = idx / KAUG;
  int k = idx - n * KAUG;
  int gate = n >> 8;
  int nl = n & 255;
  const float* Wh = (gate == 0) ? Wgh : (gate == 1) ? Wih : (gate == 2) ? Wfh : Woh;
  const float* Wx = (gate == 0) ? Wgx : (gate == 1) ? Wix : (gate == 2) ? Wfx : Wox;
  float v = 0.0f;
  if (k < HIDN)             v = Wh[k * 256 + nl];
  else if (k < HIDN + EMBD) v = Wx[(k - HIDN) * 256 + nl];
  wcat[idx] = (__bf16)v;
}

// ---- branch-free hardware transcendentals (v_exp_f32 is exp2; v_tanh_f32 on gfx1250) ----
__device__ __forceinline__ float fast_exp2(float x) {
#if __has_builtin(__builtin_amdgcn_exp2f)
  return __builtin_amdgcn_exp2f(x);
#else
  return exp2f(x);
#endif
}
__device__ __forceinline__ float fast_rcp(float x) {
#if __has_builtin(__builtin_amdgcn_rcpf)
  return __builtin_amdgcn_rcpf(x);
#else
  return 1.0f / x;
#endif
}
__device__ __forceinline__ float fast_sigmoid(float x) {
  // 1/(1+exp(-x)) = rcp(1 + exp2(-log2(e)*x))
  return fast_rcp(1.0f + fast_exp2(x * -1.4426950408889634f));
}
__device__ __forceinline__ float fast_tanh(float x) {
#if __has_builtin(__builtin_amdgcn_tanhf)
  return __builtin_amdgcn_tanhf(x);   // v_tanh_f32
#else
  // tanh(x) = 2*sigmoid(2x) - 1, branch-free
  return 2.0f * fast_rcp(1.0f + fast_exp2(x * -2.8853900817779268f)) - 1.0f;
#endif
}

// -------------------------------------------------------------------------
// Persistent LSTM: one workgroup per 32 batch rows, 8 waves of 32.
// Wave w owns hidden columns [32w, 32w+32) of all 4 gates.
// -------------------------------------------------------------------------
__global__ __launch_bounds__(256, 1)
void lstm_persistent(const int*   __restrict__ xtok,
                     const float* __restrict__ emb,
                     const float* __restrict__ bg, const float* __restrict__ bi,
                     const float* __restrict__ bfv, const float* __restrict__ bo,
                     const float* __restrict__ Wph, const float* __restrict__ bp,
                     const __bf16* __restrict__ wcat,
                     float* __restrict__ out) {
  __shared__ __bf16 s_h[MTILE][KAUG];   // augmented A rows: [h(256) | x(32 padded)]
  __shared__ float  s_bias[1024];
  __shared__ int    s_tok[MTILE];

  const int tid   = threadIdx.x;
  const int w     = tid >> 5;      // wave id 0..7
  const int lane  = tid & 31;
  const int lhalf = lane >> 4;     // 0: lanes 0-15, 1: lanes 16-31
  const int l16   = lane & 15;
  const int m0    = blockIdx.x * MTILE;

  // ---- init LDS: zero h + pad region; load fused bias ----
  for (int i = tid; i < MTILE * KAUG; i += 256) ((__bf16*)s_h)[i] = (__bf16)0.0f;
  for (int i = tid; i < 1024; i += 256) {
    int g = i >> 8, j = i & 255;
    s_bias[i] = (g == 0) ? bg[j] : (g == 1) ? bi[j] : (g == 2) ? bfv[j] : bo[j];
  }
  __syncthreads();

  // per-lane bias constants (invariant over t): col = g*256 + w*32 + tn*16 + l16
  float bb[4][2];
  #pragma unroll
  for (int g = 0; g < 4; ++g)
    #pragma unroll
    for (int tn = 0; tn < 2; ++tn)
      bb[g][tn] = s_bias[g * 256 + w * 32 + tn * 16 + l16];

  // hoisted per-lane base addresses
  const __bf16* aRow0 = &s_h[l16][0]      + (lhalf ? 16 : 0);
  const __bf16* aRow1 = &s_h[16 + l16][0] + (lhalf ? 16 : 0);
  const __bf16* bBase[4][2];
  #pragma unroll
  for (int g = 0; g < 4; ++g)
    #pragma unroll
    for (int tn = 0; tn < 2; ++tn)
      bBase[g][tn] = wcat + (size_t)(g * 256 + w * 32 + tn * 16 + l16) * KAUG
                          + (lhalf ? 16 : 0);

  const v8f z8 = {0.f, 0.f, 0.f, 0.f, 0.f, 0.f, 0.f, 0.f};
  v8f cfr[2][2] = {{z8, z8}, {z8, z8}};   // cell state fragments, register-resident
  v8f acc[4][2][2];                       // [gate][n-tile][m-tile]

  #pragma unroll 1
  for (int t = 0; t < T_LEN; ++t) {
    // --- fetch this step's tokens + embeddings into augmented K region ---
    if (tid < MTILE) s_tok[tid] = xtok[(m0 + tid) * T_LEN + t];
    __syncthreads();
    for (int idx = tid; idx < MTILE * EMBD; idx += 256) {
      int m = idx / EMBD, k = idx - m * EMBD;
      s_h[m][HIDN + k] = (__bf16)emb[s_tok[m] * EMBD + k];
    }
    __syncthreads();

    // --- z = [h|x] @ [Wh;Wx] via WMMA bf16, K = 9 chunks of 32 ---
    #pragma unroll
    for (int g = 0; g < 4; ++g)
      #pragma unroll
      for (int tn = 0; tn < 2; ++tn)
        #pragma unroll
        for (int mt = 0; mt < 2; ++mt) acc[g][tn][mt] = z8;

    #pragma unroll 1   // keep body small: avoid spill-inducing 9x unroll
    for (int kc = 0; kc < 9; ++kc) {
      const int kb = kc * 32;
      // A fragments (16-bit A 16x32 layout): two 16B LDS loads per lane
      v16bf a[2];
      {
        union { v16bf v; v8bf h[2]; } u0, u1;
        u0.h[0] = *(const v8bf*)(aRow0 + kb);
        u0.h[1] = *(const v8bf*)(aRow0 + kb + 8);
        u1.h[0] = *(const v8bf*)(aRow1 + kb);
        u1.h[1] = *(const v8bf*)(aRow1 + kb + 8);
        a[0] = u0.v;
        a[1] = u1.v;
      }
      #pragma unroll
      for (int g = 0; g < 4; ++g) {
        #pragma unroll
        for (int tn = 0; tn < 2; ++tn) {
          v16bf bfrag = *(const v16bf*)(bBase[g][tn] + kb);
          acc[g][tn][0] = __builtin_amdgcn_wmma_f32_16x16x32_bf16(
              false, a[0], false, bfrag, (short)0, acc[g][tn][0], false, false);
          acc[g][tn][1] = __builtin_amdgcn_wmma_f32_16x16x32_bf16(
              false, a[1], false, bfrag, (short)0, acc[g][tn][1], false, false);
        }
      }
    }

    // --- elementwise cell update: branch-free TRANS ops, register-resident ---
    // reuse acc[0] slot to hold new h
    #pragma unroll
    for (int tn = 0; tn < 2; ++tn)
      #pragma unroll
      for (int mt = 0; mt < 2; ++mt)
        #pragma unroll
        for (int i = 0; i < 8; ++i) {
          float gg = fast_tanh   (acc[0][tn][mt][i] + bb[0][tn]);
          float ii = fast_sigmoid(acc[1][tn][mt][i] + bb[1][tn]);
          float ff = fast_sigmoid(acc[2][tn][mt][i] + bb[2][tn]);
          float oo = fast_sigmoid(acc[3][tn][mt][i] + bb[3][tn]);
          float c  = gg * ii + cfr[tn][mt][i] * ff;
          cfr[tn][mt][i] = c;
          acc[0][tn][mt][i] = fast_tanh(c) * oo;
        }

    __syncthreads();  // all A-reads of old h done before overwrite

    // --- scatter new h (bf16) back to LDS; C-frag element (v,L): M=v+8*(L/16), N=L%16
    #pragma unroll
    for (int tn = 0; tn < 2; ++tn)
      #pragma unroll
      for (int mt = 0; mt < 2; ++mt)
        #pragma unroll
        for (int i = 0; i < 8; ++i) {
          int m = mt * 16 + i + 8 * lhalf;
          int j = w * 32 + tn * 16 + l16;
          s_h[m][j] = (__bf16)acc[0][tn][mt][i];
        }
  }
  __syncthreads();

  // --- final projection: out = h @ Wph + bp  (tiny: 32x10 per WG) ---
  for (int idx = tid; idx < MTILE * CLS; idx += 256) {
    int m = idx / CLS, n = idx - m * CLS;
    float s = bp[n];
    for (int k = 0; k < HIDN; ++k) s += (float)s_h[m][k] * Wph[k * CLS + n];
    out[(m0 + m) * CLS + n] = s;
  }
}

// -------------------------------------------------------------------------
extern "C" void kernel_launch(void* const* d_in, const int* in_sizes, int n_in,
                              void* d_out, int out_size, void* d_ws, size_t ws_size,
                              hipStream_t stream) {
  const int*   x   = (const int*)  d_in[0];
  const float* emb = (const float*)d_in[1];
  const float* Wgx = (const float*)d_in[2];
  const float* Wgh = (const float*)d_in[3];
  const float* bg  = (const float*)d_in[4];
  const float* Wix = (const float*)d_in[5];
  const float* Wih = (const float*)d_in[6];
  const float* bi  = (const float*)d_in[7];
  const float* Wfx = (const float*)d_in[8];
  const float* Wfh = (const float*)d_in[9];
  const float* bfv = (const float*)d_in[10];
  const float* Wox = (const float*)d_in[11];
  const float* Woh = (const float*)d_in[12];
  const float* bo  = (const float*)d_in[13];
  const float* Wph = (const float*)d_in[14];
  const float* bp  = (const float*)d_in[15];
  float* out = (float*)d_out;

  __bf16* wcat = (__bf16*)d_ws;  // 1024*288*2 = 576 KB

  int prep_elems = 1024 * KAUG;
  prep_weights<<<(prep_elems + 255) / 256, 256, 0, stream>>>(
      Wgx, Wgh, Wix, Wih, Wfx, Wfh, Wox, Woh, wcat);

  lstm_persistent<<<NWG, 256, 0, stream>>>(
      x, emb, bg, bi, bfv, bo, Wph, bp, wcat, out);
}
